// AngleConvCat_80917183856998
// MI455X (gfx1250) — matrix operand order repaired and checked by reading, not compiled, not run on hardware
//
#include <hip/hip_runtime.h>
#include <hip/hip_bf16.h>

// POD vector types (safe inside unions, unlike HIP's class-based float4/uint4)
typedef __attribute__((ext_vector_type(16))) _Float16     v16h;
typedef __attribute__((ext_vector_type(8)))  float        v8f;
typedef __attribute__((ext_vector_type(4)))  unsigned int u32x4;
typedef __attribute__((ext_vector_type(2)))  unsigned int u32x2;
typedef __attribute__((ext_vector_type(4)))  float        f32x4;

#define ADIM  256     // angle/output dim (= GEMM N per matrix)
#define NCOLS 512     // core(256) | gate(256) computed together
#define MT    64      // angle rows per block

union AFrag { u32x4 u[2]; v16h v; };

// ---------------------------------------------------------------------------
// Kernel 1: convert W_core/W_gate (f32, [1024][256]) into one f16 matrix
// transposed to [col 0..511][k 0..1023]; 1 MB result stays L2-resident.
// ---------------------------------------------------------------------------
__global__ void wcvt_kernel(const float* __restrict__ Wc, const float* __restrict__ Wg,
                            unsigned short* __restrict__ Wh) {
  int id = blockIdx.x * 256 + threadIdx.x;       // 0 .. 512*1024-1
  int nn = id >> 10, k = id & 1023;
  float w = (nn < ADIM) ? Wc[(size_t)k * ADIM + nn]
                        : Wg[(size_t)k * ADIM + (nn - ADIM)];
  union { _Float16 h; unsigned short s; } cv; cv.h = (_Float16)w;
  Wh[id] = cv.s;
}

// ---------------------------------------------------------------------------
// Kernel 2: fused gather + dual GEMM (core|gate) via v_wmma_f32_16x16x32_f16.
// Block: 256 threads = 8 waves; wave tile = 16 rows x 256 cols (16 accums).
// Pipeline: weights for K-chunk kc+1 are loaded to registers before the 16
// WMMAs of chunk kc (global latency hidden), B LDS double-buffered (one
// barrier per chunk), B fragments double-buffered in registers (dscnt relax).
// Per-block column sum/sumsq computed from the accumulators (deterministic).
// ---------------------------------------------------------------------------
__global__ __launch_bounds__(256) void gemm_kernel(
    const float* __restrict__ vertex, const float* __restrict__ edge,
    const float* __restrict__ angle,
    const int* __restrict__ jidx, const int* __restrict__ kidx,
    const int* __restrict__ iidx,
    const unsigned short* __restrict__ Wh,
    float* __restrict__ coreWS, float* __restrict__ gateWS,
    float* __restrict__ partial, int nAngles)
{
  __shared__ unsigned short Alds[MT * 256];          // 32 KB
  __shared__ unsigned short Blds[2 * NCOLS * 32];    // 2 x 32 KB (double buffer)
  __shared__ int   Sj[MT], Sk[MT], Si[MT];
  __shared__ float Ssum[4 * NCOLS], Ssq[4 * NCOLS];  // 16 KB

  const int  tid  = threadIdx.x;
  const int  wave = tid >> 5, lane = tid & 31;
  const int  g = lane >> 4, m = lane & 15;
  const int  half = wave >> 2, mw = wave & 3;
  const long row0 = (long)blockIdx.x * MT;

  if (tid < MT)          { long r = row0 + tid;          if (r >= nAngles) r = nAngles - 1; Sj[tid]          = jidx[r]; }
  else if (tid < 2 * MT) { long r = row0 + (tid - MT);   if (r >= nAngles) r = nAngles - 1; Sk[tid - MT]     = kidx[r]; }
  else if (tid < 3 * MT) { long r = row0 + (tid - 2*MT); if (r >= nAngles) r = nAngles - 1; Si[tid - 2*MT]   = iidx[r]; }

  v8f acc[16];
  #pragma unroll
  for (int i = 0; i < 16; ++i) acc[i] = v8f{0.f,0.f,0.f,0.f,0.f,0.f,0.f,0.f};

  const unsigned int* Wu = (const unsigned int*)Wh;  // [512 cols][512 dwords]
  // B staging: thread t copies weight rows 2t and 2t+1 (16 dwords each) as b128s
  const size_t wrow = (size_t)(2 * tid) * 512;

  for (int seg = 0; seg < 4; ++seg) {
    __syncthreads();                                 // prior compute done with Alds
    // --- stage A segment: 64 rows x 256 f32 -> f16 into LDS
    #pragma unroll 4
    for (int i = 0; i < 16; ++i) {
      int c = tid + i * 256;                         // row*64 + float4-offset
      int r = c >> 6, off = c & 63;
      const float* src;
      if (seg == 0)      src = vertex + (size_t)Sj[r] * 256;
      else if (seg == 1) src = edge   + (size_t)Sk[r] * 256;
      else if (seg == 2) src = edge   + (size_t)Si[r] * 256;
      else { long gr = row0 + r; if (gr >= nAngles) gr = nAngles - 1; src = angle + (size_t)gr * 256; }
      f32x4 f = ((const f32x4*)src)[off];
      union { _Float16 h[4]; u32x2 u; } p;
      p.h[0] = (_Float16)f.x; p.h[1] = (_Float16)f.y;
      p.h[2] = (_Float16)f.z; p.h[3] = (_Float16)f.w;
      *(u32x2*)&Alds[r * 256 + off * 4] = p.u;
    }
    const int kdw0 = seg * 128;
    // prime the register pipeline with chunk kc=0 weights
    u32x4 rg[8];
    {
      const u32x4* gs = (const u32x4*)(Wu + wrow + kdw0);
      rg[0] = gs[0];   rg[1] = gs[1];   rg[2] = gs[2];   rg[3] = gs[3];
      rg[4] = gs[128]; rg[5] = gs[129]; rg[6] = gs[130]; rg[7] = gs[131];
    }
    for (int kc = 0; kc < 8; ++kc) {
      unsigned short* Bb = Blds + (kc & 1) * (NCOLS * 32);
      u32x4* ld = (u32x4*)Bb + tid * 8;              // rows 2t,2t+1 contiguous
      ld[0] = rg[0]; ld[1] = rg[1]; ld[2] = rg[2]; ld[3] = rg[3];
      ld[4] = rg[4]; ld[5] = rg[5]; ld[6] = rg[6]; ld[7] = rg[7];
      __syncthreads();                               // chunk kc visible to all
      if (kc < 7) {                                  // issue kc+1 loads now; they
        const u32x4* gs = (const u32x4*)(Wu + wrow + kdw0 + (kc + 1) * 16);
        rg[0] = gs[0];   rg[1] = gs[1];   rg[2] = gs[2];   rg[3] = gs[3];
        rg[4] = gs[128]; rg[5] = gs[129]; rg[6] = gs[130]; rg[7] = gs[131];
      }                                              // complete under the WMMAs
      // --- compute: A frag (lane m = row; k = g*16..+15)
      AFrag a;
      const unsigned short* ap = &Alds[(mw * 16 + m) * 256 + kc * 32 + g * 16];
      a.u[0] = *(const u32x4*)(ap);
      a.u[1] = *(const u32x4*)(ap + 8);
      const unsigned short* bbase = &Bb[(half * 256 + m) * 32 + g * 16];
      AFrag b[2];
      b[0].u[0] = *(const u32x4*)(bbase);
      b[0].u[1] = *(const u32x4*)(bbase + 8);
      #pragma unroll
      for (int nt = 0; nt < 16; ++nt) {              // reg double-buffered B frags
        if (nt < 15) {
          const unsigned short* bp = bbase + (nt + 1) * 16 * 32;
          b[(nt + 1) & 1].u[0] = *(const u32x4*)(bp);
          b[(nt + 1) & 1].u[1] = *(const u32x4*)(bp + 8);
        }
        acc[nt] = __builtin_amdgcn_wmma_f32_16x16x32_f16(
            false, a.v, false, b[nt & 1].v, (short)0, acc[nt], false, false);
      }
    }
  }

  // --- store C + per-block column sum/sumsq (masking invalid tail rows)
  float* outp = half ? gateWS : coreWS;
  #pragma unroll
  for (int nt = 0; nt < 16; ++nt) {
    int col = nt * 16 + m;
    float s = 0.f, q = 0.f;
    #pragma unroll
    for (int v = 0; v < 8; ++v) {                    // C layout: M = v + 8*g, N = m
      long grow = row0 + mw * 16 + v + 8 * g;
      float x = acc[nt][v];
      if (grow < nAngles) { outp[grow * 256 + col] = x; s += x; q += x * x; }
    }
    s += __shfl_xor(s, 16, 32);
    q += __shfl_xor(q, 16, 32);
    if (g == 0) { Ssum[mw * NCOLS + half * 256 + col] = s;
                  Ssq [mw * NCOLS + half * 256 + col] = q; }
  }
  __syncthreads();
  for (int t = tid; t < NCOLS; t += 256) {           // fixed-order deterministic fold
    float s = 0.f, q = 0.f;
    #pragma unroll
    for (int w = 0; w < 4; ++w) { s += Ssum[w * NCOLS + t]; q += Ssq[w * NCOLS + t]; }
    partial[(size_t)blockIdx.x * 1024 + t]         = s;
    partial[(size_t)blockIdx.x * 1024 + NCOLS + t] = q;
  }
}

// ---------------------------------------------------------------------------
// Kernel 3: deterministic combine of per-block partials -> BN scale/shift.
// coef layout: [scale_c 256][shift_c 256][scale_g 256][shift_g 256]
// ---------------------------------------------------------------------------
__global__ void stats_combine(const float* __restrict__ partial, int nBlocks,
                              const float* __restrict__ gc, const float* __restrict__ bc,
                              const float* __restrict__ gg, const float* __restrict__ bg,
                              float* __restrict__ coef, float invN)
{
  int t = blockIdx.x * 256 + threadIdx.x;            // 0..511
  if (t >= NCOLS) return;
  float s = 0.f, q = 0.f;
  for (int b = 0; b < nBlocks; ++b) {
    s += partial[(size_t)b * 1024 + t];
    q += partial[(size_t)b * 1024 + NCOLS + t];
  }
  float mean = s * invN;
  float var  = fmaxf(q * invN - mean * mean, 0.f);
  float inv  = 1.f / sqrtf(var + 1e-5f);
  if (t < ADIM) { float sc = gc[t] * inv; coef[t] = sc;           coef[256 + t] = bc[t] - mean * sc; }
  else { int c = t - ADIM; float sc = gg[c] * inv; coef[512 + c] = sc; coef[768 + c] = bg[c] - mean * sc; }
}

// ---------------------------------------------------------------------------
// Kernel 4: fused BN + SiLU(core) * sigmoid(gate) + residual, float4 streams.
// ---------------------------------------------------------------------------
__global__ void bn_act_kernel(const float* __restrict__ core, const float* __restrict__ gate,
                              const float* __restrict__ angle, const float* __restrict__ coef,
                              float* __restrict__ out, long total4)
{
  long u = (long)blockIdx.x * 256 + threadIdx.x;
  if (u >= total4) return;
  int q4 = (int)(u & 63);
  f32x4 cv = ((const f32x4*)core )[u];
  f32x4 gv = ((const f32x4*)gate )[u];
  f32x4 av = ((const f32x4*)angle)[u];
  f32x4 sc  = ((const f32x4*)(coef      ))[q4];
  f32x4 shc = ((const f32x4*)(coef + 256))[q4];
  f32x4 sg  = ((const f32x4*)(coef + 512))[q4];
  f32x4 shg = ((const f32x4*)(coef + 768))[q4];
  f32x4 r;
  #pragma unroll
  for (int j = 0; j < 4; ++j) {
    float x = cv[j] * sc[j] + shc[j];
    float y = gv[j] * sg[j] + shg[j];
    float silu = x / (1.f + __expf(-x));
    float sig  = 1.f / (1.f + __expf(-y));
    r[j] = silu * sig + av[j];
  }
  ((f32x4*)out)[u] = r;
}

// ---------------------------------------------------------------------------
extern "C" void kernel_launch(void* const* d_in, const int* in_sizes, int n_in,
                              void* d_out, int out_size, void* d_ws, size_t ws_size,
                              hipStream_t stream) {
  const float* vertex = (const float*)d_in[0];
  const float* edge   = (const float*)d_in[1];
  const float* angle  = (const float*)d_in[2];
  /* d_in[3] = edge_index: unused by the reference computation */
  const int*   kidx   = (const int*)d_in[4];
  const int*   jidx   = (const int*)d_in[5];
  const int*   iidx   = (const int*)d_in[6];
  const float* Wc     = (const float*)d_in[7];
  const float* Wg     = (const float*)d_in[8];
  const float* gc     = (const float*)d_in[9];
  const float* bc     = (const float*)d_in[10];
  const float* gg     = (const float*)d_in[11];
  const float* bg     = (const float*)d_in[12];
  float* out = (float*)d_out;

  const int n = in_sizes[2] / ADIM;                  // N_ANGLES (300000)
  const int gblocks = (n + MT - 1) / MT;

  // workspace layout (bytes): [Wh f16 1MB][core f32][gate f32][partials][coef]
  char* wsb = (char*)d_ws;
  unsigned short* Wh = (unsigned short*)wsb;
  float* coreWS  = (float*)(wsb + (size_t)(1u << 20));
  float* gateWS  = coreWS + (size_t)n * ADIM;
  float* partial = gateWS + (size_t)n * ADIM;
  float* coef    = partial + (size_t)gblocks * 1024;

  wcvt_kernel<<<2048, 256, 0, stream>>>(Wc, Wg, Wh);
  gemm_kernel<<<gblocks, 256, 0, stream>>>(vertex, edge, angle, jidx, kidx, iidx,
                                           Wh, coreWS, gateWS, partial, n);
  stats_combine<<<2, 256, 0, stream>>>(partial, gblocks, gc, bc, gg, bg, coef,
                                       1.0f / (float)n);
  const long total4 = (long)n * (ADIM / 4);
  bn_act_kernel<<<(int)((total4 + 255) / 256), 256, 0, stream>>>(coreWS, gateWS, angle,
                                                                 coef, out, total4);
}
